// lstm_seq2seq_with_speed_52072183497258
// MI455X (gfx1250) — matrix operand order repaired and checked by reading, not compiled, not run on hardware
//
#include <hip/hip_runtime.h>

typedef __attribute__((ext_vector_type(16))) _Float16 v16h;
typedef __attribute__((ext_vector_type(8)))  float    v8f;

namespace {

constexpr int T_STEPS  = 1000;
constexpr int BATCH    = 2048;
constexpr int BM       = 16;    // batch rows per workgroup (one WMMA M-tile)
constexpr int NTHREADS = 128;   // 4 wave32s

__device__ __forceinline__ float rcp_fast(float x) {
  return __builtin_amdgcn_rcpf(x);              // v_rcp_f32
}
__device__ __forceinline__ float sigm(float x) {
  return rcp_fast(1.0f + __expf(-x));           // exp(-x)->inf => rcp->0  (safe)
}
__device__ __forceinline__ float tanh_f(float x) {
  // tanh(x) = 1 - 2/(exp(2x)+1); exp->inf => 1, exp->0 => -1  (safe both ends)
  return fmaf(-2.0f, rcp_fast(1.0f + __expf(2.0f * x)), 1.0f);
}

union AFrag { v16h v; unsigned u[8]; };
union BFrag { v16h v; uint4 q[2]; };
union CAcc  { v8f  v; float f[8]; };

// H  : true hidden size (K of the recurrent GEMM)
// HP : padded K / LDS row stride (multiple of 32 for f16 WMMA K-chunks)
// KC : HP/32 WMMA K-chunks
// NT : ceil(4H/16) N-tiles
template<int H, int HP, int KC, int NT, bool IS_DEC>
__global__ __launch_bounds__(NTHREADS)
void lstm_persistent(const float* __restrict__ x_seq,   // enc: input_batch, dec: target_batch
                     const float* __restrict__ Wih,     // (4H,1)
                     const float* __restrict__ Whh,     // (4H,H) row-major
                     const float* __restrict__ bih,
                     const float* __restrict__ bhh,
                     const float* __restrict__ speed,   // dec only
                     const float* __restrict__ linW,    // dec only (H)
                     const float* __restrict__ linb,    // dec only (1)
                     const float* __restrict__ denseW,  // dec only (1)
                     const float* __restrict__ denseb,  // dec only (1)
                     float* __restrict__ h_ws,          // enc hT out / dec h0 in (B*64)
                     float* __restrict__ c_ws,          // enc cT out / dec c0 in (B*64)
                     float* __restrict__ out)           // dec only (T*B)
{
  constexpr int NS  = NT * 16;   // padded gate-vector width
  constexpr int N4H = 4 * H;

  // Whh^T staged once as f16 in WMMA B-fragment order:
  //   fragment (tile,kc): lane L, element e  <->  K = kc*32 + (L>>4)*16 + e,
  //                                               N = tile*16 + (L&15)
  __shared__ _Float16 Bw[NT * KC * 512];
  __shared__ float    gbuf[BM * NS];     // gates, C-layout rows
  __shared__ float    cbuf[BM * HP];     // cell state (f32)
  __shared__ _Float16 hbuf[BM * HP];     // hidden as f16 (WMMA A source), zero-padded
  __shared__ float    hbuf32[BM * HP];   // hidden full precision
  __shared__ float    WihB[NS];
  __shared__ float    biasB[NS];
  __shared__ float    linWB[HP];
  __shared__ float    xbuf[BM];
  __shared__ float    obuf[BM];

  const int tid   = threadIdx.x;
  const int lane  = tid & 31;
  const int wave  = tid >> 5;
  const int half  = lane >> 4;
  const int nl    = lane & 15;
  const int mbase = blockIdx.x * BM;

  // ---------------- one-time prologue ----------------
  for (int idx = tid; idx < NT * KC * 512; idx += NTHREADS) {
    int frag = idx >> 9;
    int rem  = idx & 511;
    int L    = rem >> 4;
    int e    = rem & 15;
    int tile = frag / KC;
    int kc   = frag - tile * KC;
    int K    = kc * 32 + (L >> 4) * 16 + e;
    int N    = tile * 16 + (L & 15);
    float v  = (K < H && N < N4H) ? Whh[N * H + K] : 0.0f;  // Whh^T[K][N]
    Bw[idx]  = (_Float16)v;
  }
  for (int n = tid; n < NS; n += NTHREADS) {
    WihB[n]  = (n < N4H) ? Wih[n] : 0.0f;
    biasB[n] = (n < N4H) ? (bih[n] + bhh[n]) : 0.0f;
  }
  if (IS_DEC) {
    for (int j = tid; j < HP; j += NTHREADS)
      linWB[j] = (j < H) ? linW[j] : 0.0f;
  }
  for (int u = tid; u < BM * HP; u += NTHREADS) {
    int m = u / HP, j = u - m * HP;
    float h0 = 0.0f, c0 = 0.0f;
    if (IS_DEC) {
      int mg = mbase + m;
      if (j < 64)       { h0 = h_ws[mg * 64 + j]; c0 = c_ws[mg * 64 + j]; }
      else if (j == 64) { float sp = speed[mg] * denseW[0] + denseb[0]; h0 = sp; c0 = sp; }
    }
    cbuf[u]   = c0;
    hbuf[u]   = (_Float16)h0;
    hbuf32[u] = h0;
  }
  if (tid < BM) {
    int mg    = mbase + tid;
    xbuf[tid] = IS_DEC ? 0.0f : x_seq[mg];     // x_dec[0] = 0, x_enc[0] = input[0]
    obuf[tid] = 0.0f;
  }
  __syncthreads();

  float lb = 0.0f;
  if (IS_DEC) lb = linb[0];

  // ---------------- time loop ----------------
  #pragma unroll 1
  for (int t = 0; t < T_STEPS; ++t) {
    // drain previous step's projected output (same threads that zero obuf)
    if (IS_DEC) {
      if (t > 0 && tid < BM) {
        out[(t - 1) * BATCH + (mbase + tid)] = obuf[tid] + lb;
        obuf[tid] = 0.0f;
      }
    }

    // -- phase A: gates = h @ Whh^T + x*Wih + b   (WMMA) --
    float xv[8];
    #pragma unroll
    for (int v = 0; v < 8; ++v) xv[v] = xbuf[v + 8 * half];   // C-layout M rows of this lane

    AFrag a[KC];
    #pragma unroll
    for (int kc = 0; kc < KC; ++kc) {
      #pragma unroll
      for (int v = 0; v < 8; ++v) {
        // A operand in this code's contiguous K order (matches Bw): VGPR v: K=16*half+2v(+kc*32)
        int kb = kc * 32 + 16 * half + 2 * v;
        a[kc].u[v] = *(const unsigned*)&hbuf[nl * HP + kb];   // packed pair (K, K+1)
      }
    }

    for (int tile = wave; tile < NT; tile += NTHREADS / 32) {
      int   n   = tile * 16 + nl;
      float wih = WihB[n], bb = biasB[n];
      CAcc acc;
      #pragma unroll
      for (int v = 0; v < 8; ++v) acc.f[v] = xv[v] * wih + bb;
      #pragma unroll
      for (int kc = 0; kc < KC; ++kc) {
        BFrag b;
        const _Float16* bp = &Bw[(tile * KC + kc) * 512 + lane * 16];
        b.q[0] = *(const uint4*)(bp);
        b.q[1] = *(const uint4*)(bp + 8);
        acc.v  = __builtin_amdgcn_wmma_f32_16x16x32_f16(
                     false, a[kc].v, false, b.v, (short)0, acc.v, false, false);
      }
      #pragma unroll
      for (int v = 0; v < 8; ++v)
        gbuf[(v + 8 * half) * NS + n] = acc.f[v];   // C layout: M = v + 8*half, N = n
    }
    __syncthreads();

    // -- phase B: elementwise LSTM cell update (PyTorch gate order i,f,g,o) --
    for (int u = tid; u < BM * H; u += NTHREADS) {
      int m = u / H, j = u - m * H;
      const float* gr = &gbuf[m * NS];
      float gi = gr[j], gf = gr[H + j], gg = gr[2 * H + j], go = gr[3 * H + j];
      float c  = cbuf[m * HP + j];
      c        = sigm(gf) * c + sigm(gi) * tanh_f(gg);
      float h  = sigm(go) * tanh_f(c);
      cbuf[m * HP + j]   = c;
      hbuf[m * HP + j]   = (_Float16)h;
      hbuf32[m * HP + j] = h;
      if (IS_DEC) atomicAdd(&obuf[m], h * linWB[j]);  // ds_add_f32
    }
    // stage next step's input
    if (tid < BM) {
      int mg = mbase + tid;
      if (IS_DEC)                xbuf[tid] = x_seq[t * BATCH + mg];       // x_{t+1}=target[t]
      else if (t + 1 < T_STEPS)  xbuf[tid] = x_seq[(t + 1) * BATCH + mg];
    }
    __syncthreads();
  }

  // ---------------- epilogue ----------------
  if (IS_DEC) {
    if (tid < BM)
      out[(T_STEPS - 1) * BATCH + (mbase + tid)] = obuf[tid] + lb;
  } else {
    for (int u = tid; u < BM * 64; u += NTHREADS) {
      int m = u / 64, j = u - m * 64;
      int mg = mbase + m;
      h_ws[mg * 64 + j] = hbuf32[m * HP + j];
      c_ws[mg * 64 + j] = cbuf[m * HP + j];
    }
  }
}

} // namespace

extern "C" void kernel_launch(void* const* d_in, const int* in_sizes, int n_in,
                              void* d_out, int out_size, void* d_ws, size_t ws_size,
                              hipStream_t stream) {
  const float* input  = (const float*)d_in[0];
  const float* speed  = (const float*)d_in[1];
  const float* target = (const float*)d_in[2];
  const float* eWih   = (const float*)d_in[3];
  const float* eWhh   = (const float*)d_in[4];
  const float* ebih   = (const float*)d_in[5];
  const float* ebhh   = (const float*)d_in[6];
  const float* dWih   = (const float*)d_in[7];
  const float* dWhh   = (const float*)d_in[8];
  const float* dbih   = (const float*)d_in[9];
  const float* dbhh   = (const float*)d_in[10];
  const float* linW   = (const float*)d_in[11];
  const float* linb   = (const float*)d_in[12];
  const float* denW   = (const float*)d_in[13];
  const float* denb   = (const float*)d_in[14];
  float* out  = (float*)d_out;
  float* h_ws = (float*)d_ws;                 // B*64 f32
  float* c_ws = h_ws + BATCH * 64;            // B*64 f32

  dim3 grid(BATCH / BM), block(NTHREADS);
  // encoder: H=64, Kpad=64 (2 chunks), 4H=256 -> 16 N-tiles
  lstm_persistent<64, 64, 2, 16, false><<<grid, block, 0, stream>>>(
      input, eWih, eWhh, ebih, ebhh,
      nullptr, nullptr, nullptr, nullptr, nullptr,
      h_ws, c_ws, nullptr);
  // decoder: H=65, Kpad=96 (3 chunks), 4H=260 -> 17 N-tiles (N padded to 272)
  lstm_persistent<65, 96, 3, 17, true><<<grid, block, 0, stream>>>(
      target, dWih, dWhh, dbih, dbhh,
      speed, linW, linb, denW, denb,
      h_ws, c_ws, out);
}
